// GraphMatchingNetwork_34857954574812
// MI455X (gfx1250) — compile-verified, matrix operand order checked by reading, not run on hardware
//
#include <hip/hip_runtime.h>
#include <hip/hip_bf16.h>

typedef __attribute__((ext_vector_type(16))) __bf16 v16bf;
typedef __attribute__((ext_vector_type(8)))  float  v8f;
typedef __attribute__((ext_vector_type(4)))  unsigned int u32x4;
typedef __attribute__((ext_vector_type(8)))  int    i32x8;
typedef __attribute__((ext_vector_type(4)))  int    i32x4;

#define EPSV 1e-05f

// ---------- helpers ----------

__device__ __forceinline__ unsigned short f2bf(float f) {
  unsigned u = __builtin_bit_cast(unsigned, f);
  unsigned r = u + 0x7FFFu + ((u >> 16) & 1u);
  return (unsigned short)(r >> 16);
}

// Fragment load from a row-major bf16 tile stored as [row][k] with row-stride ld.
// Used for A fragments (row = M) and for B fragments when the tile is stored [N][K].
// CDNA5 16-bit layout: lane(0..15)->row, element j -> K = k0 + hi*8 + (j<8 ? j : 8+j)
__device__ __forceinline__ v16bf frag_rowmajor(const unsigned short* t, int row0,
                                               int ld, int k0, int hi, int ln) {
  const unsigned short* p = t + (row0 + ln) * ld + k0 + hi * 8;
  v16bf f;
#pragma unroll
  for (int j = 0; j < 8; ++j) f[j]     = __builtin_bit_cast(__bf16, p[j]);
#pragma unroll
  for (int j = 0; j < 8; ++j) f[8 + j] = __builtin_bit_cast(__bf16, p[16 + j]);
  return f;
}

// B fragment from a tile stored [K][N] with row-stride ld (e.g. V in attention).
__device__ __forceinline__ v16bf frag_colmajor(const unsigned short* t, int k0,
                                               int n0, int ld, int hi, int ln) {
  const unsigned short* p = t + (k0 + hi * 8) * ld + n0 + ln;
  v16bf f;
#pragma unroll
  for (int j = 0; j < 8; ++j) f[j]     = __builtin_bit_cast(__bf16, p[j * ld]);
#pragma unroll
  for (int j = 0; j < 8; ++j) f[8 + j] = __builtin_bit_cast(__bf16, p[(16 + j) * ld]);
  return f;
}

#define WMMA_BF16(a, b, c) \
  __builtin_amdgcn_wmma_f32_16x16x32_bf16(false, (a), false, (b), (short)0, (c), false, false)

// ---------- Tensor Data Mover: 2D f32 tile (rows x dim0) global -> LDS ----------
// Builds D# group0/group1 per CDNA5 ISA 8.3/8.4; groups 2/3 zero (2D tensor).
// This toolchain exposes the 6-arg builtin form.
__device__ __forceinline__ void tdm_load_2d_f32(unsigned lds_off, const void* gptr,
                                                unsigned dim0, unsigned rows,
                                                unsigned stride_elems) {
  unsigned long long ga = (unsigned long long)gptr;
  u32x4 g0;
  g0[0] = 1u;                                           // count=1 (valid), user mode
  g0[1] = lds_off;                                      // lds_addr [63:32]
  g0[2] = (unsigned)(ga & 0xFFFFFFFFull);               // global_addr[31:0] -> bits 95:64
  g0[3] = (unsigned)((ga >> 32) & 0x01FFFFFFull)        // global_addr[56:32]
        | (2u << 30);                                   // type=2 ("image")
  i32x8 g1;
  g1[0] = (int)(2u << 16);                              // data_size=2 (4 bytes), mask=0
  g1[1] = (int)((dim0 & 0xFFFFu) << 16);                // tensor_dim0[15:0] @ bits 63:48
  g1[2] = (int)(((dim0 >> 16) & 0xFFFFu)                // tensor_dim0[31:16]
        | ((rows & 0xFFFFu) << 16));                    // tensor_dim1[15:0]
  g1[3] = (int)(((rows >> 16) & 0xFFFFu)                // tensor_dim1[31:16]
        | ((dim0 & 0xFFFFu) << 16));                    // tile_dim0
  g1[4] = (int)(rows & 0xFFFFu);                        // tile_dim1 (tile_dim2=0)
  g1[5] = (int)stride_elems;                            // tensor_dim0_stride[31:0]
  g1[6] = 0;                                            // stride hi / dim1_stride lo
  g1[7] = 0;
  i32x4 z4 = {0, 0, 0, 0};
  i32x8 z8 = {0, 0, 0, 0, 0, 0, 0, 0};
  __builtin_amdgcn_tensor_load_to_lds(g0, g1, z4, z4, z8, 0);
}

// ---------- generic GEMM: C[M,128] = A[M,K] @ B[K,128] + bias ----------
// block = 256 threads (8 waves), tile 128x128, K in chunks of 32.
// B chunk staged via TDM (tensor_load_to_lds) then converted to bf16 [N][K].
__global__ __launch_bounds__(256) void gemm128_bias(
    const float* __restrict__ A, const float* __restrict__ B,
    const float* __restrict__ bias, float* __restrict__ C,
    int K, int lda, int ldc) {
  __shared__ unsigned short As[128 * 32];
  __shared__ unsigned short Bs[128 * 32];  // stored [N][K]
  __shared__ float Bstage[32 * 128];       // TDM destination, [k][n] f32
  const int tid = threadIdx.x;
  const int lane = tid & 31, wave = tid >> 5;
  const int hi = (lane >> 4) & 1, ln = lane & 15;
  const int rowBase = blockIdx.x * 128;
  const int wm = (wave & 3) * 32, wn = (wave >> 2) * 64;
  const unsigned bstageOff = (unsigned)(unsigned long long)(void*)Bstage;
  v8f acc[2][4] = {};

  for (int k0 = 0; k0 < K; k0 += 32) {
    // async DMA of the 32x128 f32 weight chunk into LDS (one wave issues)
    if (wave == 0) {
      tdm_load_2d_f32(bstageOff, B + (size_t)k0 * 128, 128u, 32u, 128u);
      __builtin_amdgcn_s_wait_tensorcnt(0);
    }
    // stage A as bf16 concurrently with the DMA
    const int m = tid >> 1;
    const int kk = (tid & 1) * 16;
    const float* ap = A + (size_t)(rowBase + m) * lda + k0 + kk;
    __builtin_prefetch(ap + 32, 0, 1);
#pragma unroll
    for (int j = 0; j < 16; ++j) As[m * 32 + kk + j] = f2bf(ap[j]);
    __syncthreads();
    // convert TDM-staged B chunk to bf16 [n][k]
#pragma unroll
    for (int j = 0; j < 16; ++j) Bs[m * 32 + kk + j] = f2bf(Bstage[(kk + j) * 128 + m]);
    __syncthreads();
#pragma unroll
    for (int mi = 0; mi < 2; ++mi) {
      v16bf a = frag_rowmajor(As, wm + mi * 16, 32, 0, hi, ln);
#pragma unroll
      for (int ni = 0; ni < 4; ++ni) {
        v16bf b = frag_rowmajor(Bs, wn + ni * 16, 32, 0, hi, ln);
        acc[mi][ni] = WMMA_BF16(a, b, acc[mi][ni]);
      }
    }
    __syncthreads();
  }
#pragma unroll
  for (int mi = 0; mi < 2; ++mi)
#pragma unroll
    for (int ni = 0; ni < 4; ++ni)
#pragma unroll
      for (int v = 0; v < 8; ++v) {
        int row = rowBase + wm + mi * 16 + v + 8 * hi;
        int col = wn + ni * 16 + ln;
        C[(size_t)row * ldc + col] = acc[mi][ni][v] + bias[col];
      }
}

// ---------- edge message GEMM fused with segment-sum scatter ----------
// Per edge e: m_e = [h[dst_e] | h[src_e]] @ Wm + bm ; aggr[dst_e] += m_e
// aggrC is the [N,256] concat buffer; messages land in columns 0..127.
__global__ __launch_bounds__(256) void edge_gemm_scatter(
    const float* __restrict__ H, const int* __restrict__ srcI,
    const int* __restrict__ dstI, const float* __restrict__ Wm,
    const float* __restrict__ bm, float* __restrict__ aggrC, int nodeBase) {
  __shared__ unsigned short As[128 * 32];
  __shared__ unsigned short Bs[128 * 32];
  __shared__ float Bstage[32 * 128];
  __shared__ int dL[128];
  __shared__ int sL[128];
  const int tid = threadIdx.x;
  const int lane = tid & 31, wave = tid >> 5;
  const int hi = (lane >> 4) & 1, ln = lane & 15;
  const int eBase = blockIdx.x * 128;
  const int wm = (wave & 3) * 32, wn = (wave >> 2) * 64;
  const unsigned bstageOff = (unsigned)(unsigned long long)(void*)Bstage;

  if (tid < 128) dL[tid] = dstI[eBase + tid] + nodeBase;
  else           sL[tid - 128] = srcI[eBase + tid - 128] + nodeBase;
  __syncthreads();

  v8f acc[2][4] = {};
  for (int k0 = 0; k0 < 256; k0 += 32) {
    if (wave == 0) {
      tdm_load_2d_f32(bstageOff, Wm + (size_t)k0 * 128, 128u, 32u, 128u);
      __builtin_amdgcn_s_wait_tensorcnt(0);
    }
    const int m = tid >> 1;
    const int kk = (tid & 1) * 16;
#pragma unroll
    for (int j = 0; j < 16; ++j) {
      int k = k0 + kk + j;
      int node = (k < 128) ? dL[m] : sL[m];
      int kc = (k < 128) ? k : (k - 128);
      As[m * 32 + kk + j] = f2bf(H[(size_t)node * 128 + kc]);
    }
    __syncthreads();
#pragma unroll
    for (int j = 0; j < 16; ++j) Bs[m * 32 + kk + j] = f2bf(Bstage[(kk + j) * 128 + m]);
    __syncthreads();
#pragma unroll
    for (int mi = 0; mi < 2; ++mi) {
      v16bf a = frag_rowmajor(As, wm + mi * 16, 32, 0, hi, ln);
#pragma unroll
      for (int ni = 0; ni < 4; ++ni) {
        v16bf b = frag_rowmajor(Bs, wn + ni * 16, 32, 0, hi, ln);
        acc[mi][ni] = WMMA_BF16(a, b, acc[mi][ni]);
      }
    }
    __syncthreads();
  }
#pragma unroll
  for (int mi = 0; mi < 2; ++mi)
#pragma unroll
    for (int ni = 0; ni < 4; ++ni)
#pragma unroll
      for (int v = 0; v < 8; ++v) {
        int rloc = wm + mi * 16 + v + 8 * hi;
        int col = wn + ni * 16 + ln;
        float val = acc[mi][ni][v] + bm[col];
        atomicAdd(&aggrC[(size_t)dL[rloc] * 256 + col], val);
      }
}

// ---------- flash attention (raw logits, no scale): O = softmax_row(Q@KV^T) @ KV ----------
// Serves att_x (Q=gx, KV=gy) and att_y (Q=gy, KV=gx). Block handles 128 Q rows.
// K and V are the same matrix here, so one LDS tile serves both gemms
// (read [N][K]-wise for S, [K][N]-wise for P@V).
__global__ __launch_bounds__(256) void flash_attn(
    const float* __restrict__ Q, const float* __restrict__ KV,
    float* __restrict__ Out, int nKV) {
  __shared__ unsigned short Qs[128 * 128];  // [qRow][d]
  __shared__ unsigned short KVs[32 * 128];  // [kv][d]
  __shared__ unsigned short Ps[8 * 16 * 32];
  const int tid = threadIdx.x;
  const int lane = tid & 31, wave = tid >> 5;
  const int hi = (lane >> 4) & 1, ln = lane & 15;
  const int qBase = blockIdx.x * 128;

  {  // stage Q tile as bf16
    int m = tid >> 1, c0 = (tid & 1) * 64;
    const float* qp = Q + (size_t)(qBase + m) * 128 + c0;
#pragma unroll
    for (int j = 0; j < 64; ++j) Qs[m * 128 + c0 + j] = f2bf(qp[j]);
  }
  v8f o[8] = {};
  float mrow[8], lrow[8];
#pragma unroll
  for (int v = 0; v < 8; ++v) { mrow[v] = -3.0e38f; lrow[v] = 0.f; }
  __syncthreads();

  for (int kv0 = 0; kv0 < nKV; kv0 += 32) {
    {  // stage 32 K/V rows (shared tile)
      int r = tid >> 3, c0 = (tid & 7) * 16;
      const float* kp = KV + (size_t)(kv0 + r) * 128 + c0;
      __builtin_prefetch(kp + 32 * 128, 0, 1);
#pragma unroll
      for (int j = 0; j < 16; ++j) KVs[r * 128 + c0 + j] = f2bf(kp[j]);
    }
    __syncthreads();

    // S(16x32) = Qwave(16x128) @ K^T
    v8f s0 = {}, s1 = {};
#pragma unroll
    for (int k0 = 0; k0 < 128; k0 += 32) {
      v16bf a = frag_rowmajor(Qs, wave * 16, 128, k0, hi, ln);
      v16bf b0 = frag_rowmajor(KVs, 0, 128, k0, hi, ln);
      s0 = WMMA_BF16(a, b0, s0);
      v16bf b1 = frag_rowmajor(KVs, 16, 128, k0, hi, ln);
      s1 = WMMA_BF16(a, b1, s1);
    }

    // online softmax per row (row v+8*hi lives across one 16-lane half)
    unsigned short* ps = Ps + wave * 16 * 32;
#pragma unroll
    for (int v = 0; v < 8; ++v) {
      float sm = fmaxf(s0[v], s1[v]);
#pragma unroll
      for (int off = 1; off < 16; off <<= 1) sm = fmaxf(sm, __shfl_xor(sm, off, 32));
      float mn = fmaxf(mrow[v], sm);
      float alpha = __expf(mrow[v] - mn);
      mrow[v] = mn;
      float e0 = __expf(s0[v] - mn);
      float e1 = __expf(s1[v] - mn);
      float rs = e0 + e1;
#pragma unroll
      for (int off = 1; off < 16; off <<= 1) rs += __shfl_xor(rs, off, 32);
      lrow[v] = lrow[v] * alpha + rs;
#pragma unroll
      for (int d = 0; d < 8; ++d) o[d][v] *= alpha;
      int row = v + 8 * hi;
      ps[row * 32 + ln]      = f2bf(e0);
      ps[row * 32 + 16 + ln] = f2bf(e1);
    }

    // O += P(16x32) @ V(32x128)
    v16bf ap = frag_rowmajor(ps, 0, 32, 0, hi, ln);
#pragma unroll
    for (int d = 0; d < 8; ++d) {
      v16bf bv = frag_colmajor(KVs, 0, d * 16, 128, hi, ln);
      o[d] = WMMA_BF16(ap, bv, o[d]);
    }
    __syncthreads();
  }
#pragma unroll
  for (int d = 0; d < 8; ++d)
#pragma unroll
    for (int v = 0; v < 8; ++v) {
      int row = qBase + wave * 16 + v + 8 * hi;
      Out[(size_t)row * 128 + d * 16 + ln] = o[d][v] / lrow[v];
    }
}

// ---------- small elementwise / reduction kernels ----------

__global__ void zero_f32(float* p, int n) {
  int i = blockIdx.x * 256 + threadIdx.x;
  if (i < n) p[i] = 0.f;
}

// concat[:,128:256] = x - cross
__global__ void attin_kernel(const float* __restrict__ x, const float* __restrict__ cross,
                             float* __restrict__ concat, int n) {
  int i = blockIdx.x * 256 + threadIdx.x;
  if (i < n) {
    int r = i >> 7, c = i & 127;
    concat[(size_t)r * 256 + 128 + c] = x[i] - cross[i];
  }
}

// per-column sum / sumsq over all rows -> stats[0:128]=sum, stats[128:256]=sumsq
__global__ __launch_bounds__(256) void colstats_kernel(const float* __restrict__ X,
                                                       float* __restrict__ stats, int Nrows) {
  __shared__ float ssum[128];
  __shared__ float ssq[128];
  int t = threadIdx.x;
  if (t < 128) { ssum[t] = 0.f; ssq[t] = 0.f; }
  __syncthreads();
  int col = t & 127;
  float s = 0.f, q = 0.f;
  for (int r = blockIdx.x * 2 + (t >> 7); r < Nrows; r += gridDim.x * 2) {
    float v = X[(size_t)r * 128 + col];
    s += v; q += v * v;
  }
  atomicAdd(&ssum[col], s);
  atomicAdd(&ssq[col], q);
  __syncthreads();
  if (t < 128) {
    atomicAdd(&stats[t], ssum[t]);
    atomicAdd(&stats[128 + t], ssq[t]);
  }
}

// x = (out - mu) * rsqrt(var+eps) * gamma + beta   (mu/var per column)
__global__ void norm_kernel(const float* __restrict__ X, const float* __restrict__ stats,
                            const float* __restrict__ gamma, const float* __restrict__ beta,
                            float* __restrict__ Y, int Nrows) {
  int i = blockIdx.x * 256 + threadIdx.x;
  int n = Nrows * 128;
  if (i < n) {
    int c = i & 127;
    float invN = 1.0f / (float)Nrows;
    float mu = stats[c] * invN;
    float var = stats[128 + c] * invN - mu * mu;
    Y[i] = (X[i] - mu) * rsqrtf(var + EPSV) * gamma[c] + beta[c];
  }
}

// per-row softmax(gate logits) * states, atomically pooled per graph
__global__ __launch_bounds__(128) void gate_pool_kernel(const float* __restrict__ states,
                                                        const float* __restrict__ logits,
                                                        float* __restrict__ gsum, int N1) {
  __shared__ float red[128];
  int r = blockIdx.x, t = threadIdx.x;
  float v = logits[(size_t)r * 128 + t];
  red[t] = v; __syncthreads();
  for (int s = 64; s > 0; s >>= 1) { if (t < s) red[t] = fmaxf(red[t], red[t + s]); __syncthreads(); }
  float mx = red[0]; __syncthreads();
  float e = __expf(v - mx);
  red[t] = e; __syncthreads();
  for (int s = 64; s > 0; s >>= 1) { if (t < s) red[t] += red[t + s]; __syncthreads(); }
  float g = (e / red[0]) * states[(size_t)r * 128 + t];
  int graph = (r >= N1) ? 1 : 0;
  atomicAdd(&gsum[graph * 128 + t], g);
}

// emb[i] = (gsum[i]/count_i) @ Wf + bf
__global__ __launch_bounds__(256) void head_kernel(const float* __restrict__ gsum,
                                                   const float* __restrict__ Wf,
                                                   const float* __restrict__ bf_,
                                                   float* __restrict__ out, int N1, int N2) {
  int t = threadIdx.x;
  int i = t >> 7, j = t & 127;
  float cnt = (i == 0) ? (float)N1 : (float)N2;
  float acc = bf_[j];
#pragma unroll 8
  for (int k = 0; k < 128; ++k) acc += (gsum[i * 128 + k] / cnt) * Wf[k * 128 + j];
  out[i * 128 + j] = acc;
}

// ---------- host launcher ----------

extern "C" void kernel_launch(void* const* d_in, const int* in_sizes, int n_in,
                              void* d_out, int out_size, void* d_ws, size_t ws_size,
                              hipStream_t stream) {
  const float* feats1 = (const float*)d_in[0];
  const int*   ei1    = (const int*)d_in[1];
  const float* feats2 = (const float*)d_in[2];
  const int*   ei2    = (const int*)d_in[3];
  const float* Wn = (const float*)d_in[6];
  const float* bn = (const float*)d_in[7];
  const float* Wm = (const float*)d_in[8];
  const float* bm = (const float*)d_in[9];
  const float* Wp = (const float*)d_in[10];
  const float* bp = (const float*)d_in[11];
  const float* gamma = (const float*)d_in[12];
  const float* beta  = (const float*)d_in[13];
  const float* Wa = (const float*)d_in[14];
  const float* ba = (const float*)d_in[15];
  const float* Wg = (const float*)d_in[16];
  const float* bg = (const float*)d_in[17];
  const float* Wf = (const float*)d_in[18];
  const float* bf_ = (const float*)d_in[19];

  const int N1 = in_sizes[0] / 128;
  const int N2 = in_sizes[2] / 128;
  const int E1 = in_sizes[1] / 2;
  const int E2 = in_sizes[3] / 2;
  const int N  = N1 + N2;
  const size_t ND = (size_t)N * 128;

  float* ws = (float*)d_ws;
  float* xbuf      = ws;                 // [N,128] current features
  float* hbuf      = ws + ND;            // [N,128] h / out / states
  float* crossbuf  = ws + 2 * ND;        // [N,128] attention out / gate logits
  float* concatbuf = ws + 3 * ND;        // [N,256] aggr | (x - cross)
  float* stats     = ws + 5 * ND;        // 256 floats
  float* gsum      = stats + 256;        // 256 floats

  // x = concat(feats_1, feats_2)
  (void)hipMemcpyAsync(xbuf, feats1, (size_t)N1 * 128 * sizeof(float),
                       hipMemcpyDeviceToDevice, stream);
  (void)hipMemcpyAsync(xbuf + (size_t)N1 * 128, feats2, (size_t)N2 * 128 * sizeof(float),
                       hipMemcpyDeviceToDevice, stream);

  const int L = 3;
  for (int l = 0; l < L; ++l) {
    const float* Wnl = Wn + (size_t)l * 128 * 128;
    const float* bnl = bn + (size_t)l * 128;
    const float* Wml = Wm + (size_t)l * 256 * 128;
    const float* bml = bm + (size_t)l * 128;
    const float* Wpl = Wp + (size_t)l * 256 * 128;
    const float* bpl = bp + (size_t)l * 128;

    // h = x @ Wn + bn
    gemm128_bias<<<dim3(N / 128), 256, 0, stream>>>(xbuf, Wnl, bnl, hbuf, 128, 128, 128);

    // aggr (cols 0..127 of concat) starts at zero; attn part overwritten later
    zero_f32<<<dim3((int)((2 * ND + 255) / 256)), 256, 0, stream>>>(concatbuf, (int)(2 * ND));

    // fused edge GEMM + segment_sum (src = ei[0:E], dst = ei[E:2E])
    edge_gemm_scatter<<<dim3(E1 / 128), 256, 0, stream>>>(hbuf, ei1, ei1 + E1, Wml, bml,
                                                          concatbuf, 0);
    edge_gemm_scatter<<<dim3(E2 / 128), 256, 0, stream>>>(hbuf, ei2, ei2 + E2, Wml, bml,
                                                          concatbuf, N1);

    // cross attention: att_x = flash(gx, gy), att_y = flash(gy, gx)
    flash_attn<<<dim3(N1 / 128), 256, 0, stream>>>(xbuf, xbuf + (size_t)N1 * 128,
                                                   crossbuf, N2);
    flash_attn<<<dim3(N2 / 128), 256, 0, stream>>>(xbuf + (size_t)N1 * 128, xbuf,
                                                   crossbuf + (size_t)N1 * 128, N1);

    // concat[:,128:256] = x - cross
    attin_kernel<<<dim3((int)((ND + 255) / 256)), 256, 0, stream>>>(xbuf, crossbuf,
                                                                    concatbuf, (int)ND);

    // out = concat @ Wp + bp   (K=256)
    gemm128_bias<<<dim3(N / 128), 256, 0, stream>>>(concatbuf, Wpl, bpl, hbuf, 256, 256, 128);

    // axis-0 normalization
    zero_f32<<<dim3(1), 256, 0, stream>>>(stats, 256);
    colstats_kernel<<<dim3(96), 256, 0, stream>>>(hbuf, stats, N);
    norm_kernel<<<dim3((int)((ND + 255) / 256)), 256, 0, stream>>>(
        hbuf, stats, gamma + (size_t)l * 128, beta + (size_t)l * 128, xbuf, N);
  }

  // head: states, gate logits, gated pooling, final projection
  gemm128_bias<<<dim3(N / 128), 256, 0, stream>>>(xbuf, Wa, ba, hbuf, 128, 128, 128);
  gemm128_bias<<<dim3(N / 128), 256, 0, stream>>>(xbuf, Wg, bg, crossbuf, 128, 128, 128);
  zero_f32<<<dim3(1), 256, 0, stream>>>(gsum, 256);
  gate_pool_kernel<<<dim3(N), 128, 0, stream>>>(hbuf, crossbuf, gsum, N1);
  head_kernel<<<dim3(1), 256, 0, stream>>>(gsum, Wf, bf_, (float*)d_out, N1, N2);
}
